// EnergyTaskHeads_77592879169754
// MI455X (gfx1250) — matrix operand order repaired and checked by reading, not compiled, not run on hardware
//
#include <hip/hip_runtime.h>
#include <hip/hip_bf16.h>

typedef __attribute__((ext_vector_type(16))) _Float16 v16h;
typedef __attribute__((ext_vector_type(8)))  _Float16 v8h;
typedef __attribute__((ext_vector_type(4)))  _Float16 v4h;
typedef __attribute__((ext_vector_type(8)))  float    v8f;
typedef __attribute__((ext_vector_type(4)))  float    v4f;

#define NBLD 8192
#define DMOD 128
#define DHALF 64
#define SGRP 64
#define NGRP 128
#define KCL 4
#define KM_ITERS 10
#define EPSF 1e-6f

#define APAD 136   // half elements per padded row (272B: 16B-aligned, bank-advance 4)
#define HPAD 129   // float elements per padded Hs row (conflict-free column reads)

// ---------------------------------------------------------------------------
// Kernel 1: fused encoder.  One workgroup = one LV group (64 rows).
//   H = X@W1 + b1 ; LayerNorm ; ReLU ; F = H@W2 + b2
// v_wmma_f32_16x16x32_f16; A/B fragments loaded as 2x ds_load_b128 each.
// Weights are stored TRANSPOSED in LDS (Wt[N][K]) so B fragments are K-unit-
// stride per lane (B layout: K = e + 16*laneHalf, N = lane%16).
// ---------------------------------------------------------------------------
__global__ __launch_bounds__(128) void enc_kernel(
    const float* __restrict__ X,      // [8192,128]
    const float* __restrict__ W1,     // [128,128]
    const float* __restrict__ b1,
    const float* __restrict__ gma,
    const float* __restrict__ bta,
    const float* __restrict__ W2,     // [128,64]
    const float* __restrict__ b2,
    float* __restrict__ feats)        // [8192,64]
{
    __shared__ __align__(16) _Float16 Ah[SGRP][APAD];   // X tile / normalized H
    __shared__ __align__(16) _Float16 W1t[DMOD][APAD];  // W1^T: [N][K]
    __shared__ __align__(16) _Float16 W2t[DHALF][APAD]; // W2^T: [N][K]
    __shared__ float Hs[SGRP][HPAD];
    __shared__ float b1s[DMOD], gs[DMOD], bs[DMOD], b2s[DHALF];

    const int t    = threadIdx.x;
    const int g    = blockIdx.x;
    const int wave = t >> 5;
    const int lane = t & 31;
    const int mrow = lane & 15;
    const int lh   = lane >> 4;            // lane half (0/1)

    // ---- cooperative loads (float4 vmem) ----
    const float4* Xv = (const float4*)(X + (size_t)g * SGRP * DMOD);
    for (int i = t; i < SGRP * DMOD / 4; i += 128) {
        float4 x = Xv[i];
        const int r = i >> 5, c = (i & 31) * 4;
        v4h h; h[0] = (_Float16)x.x; h[1] = (_Float16)x.y;
               h[2] = (_Float16)x.z; h[3] = (_Float16)x.w;
        *(v4h*)&Ah[r][c] = h;
    }
    const float4* W1v = (const float4*)W1;
    for (int i = t; i < DMOD * DMOD / 4; i += 128) {
        float4 w = W1v[i];
        const int k = i >> 5, n = (i & 31) * 4;   // transpose: Wt[n][k]
        W1t[n + 0][k] = (_Float16)w.x; W1t[n + 1][k] = (_Float16)w.y;
        W1t[n + 2][k] = (_Float16)w.z; W1t[n + 3][k] = (_Float16)w.w;
    }
    const float4* W2v = (const float4*)W2;
    for (int i = t; i < DMOD * DHALF / 4; i += 128) {
        float4 w = W2v[i];
        const int k = i >> 4, n = (i & 15) * 4;
        W2t[n + 0][k] = (_Float16)w.x; W2t[n + 1][k] = (_Float16)w.y;
        W2t[n + 2][k] = (_Float16)w.z; W2t[n + 3][k] = (_Float16)w.w;
    }
    b1s[t] = b1[t]; gs[t] = gma[t]; bs[t] = bta[t];
    if (t < DHALF) b2s[t] = b2[t];
    __syncthreads();

    // ---- GEMM1: H[64,128] = X @ W1 + b1 ----
    {
        const int M = wave * 16 + mrow;
        const v8h* Arow = (const v8h*)&Ah[M][0];      // 17 v8h per row
        for (int n = 0; n < 8; ++n) {
            const int N = n * 16 + mrow;
            const v8h* Brow = (const v8h*)&W1t[N][0];
            v8f c = {};
            for (int kt = 0; kt < 4; ++kt) {
                // A: K = kt*32 + 16*lh + {0..15}
                v8h alo = Arow[kt * 4 + 2 * lh];
                v8h ahi = Arow[kt * 4 + 2 * lh + 1];
                // B: K = kt*32 + 16*lh + {0..15}
                v8h blo = Brow[kt * 4 + 2 * lh];
                v8h bhi = Brow[kt * 4 + 2 * lh + 1];
                v16h a, b;
#pragma unroll
                for (int e = 0; e < 8; ++e) {
                    a[e] = alo[e]; a[e + 8] = ahi[e];
                    b[e] = blo[e]; b[e + 8] = bhi[e];
                }
                c = __builtin_amdgcn_wmma_f32_16x16x32_f16(
                        false, a, false, b, (short)0, c, false, false);
            }
#pragma unroll
            for (int r = 0; r < 8; ++r) {
                const int Mo = wave * 16 + r + (lh << 3);
                const int No = n * 16 + mrow;
                Hs[Mo][No] = c[r] + b1s[No];
            }
        }
    }
    __syncthreads();

    // ---- LayerNorm + ReLU (one thread per row) -> half into Ah ----
    if (t < SGRP) {
        float mu = 0.f;
        for (int j = 0; j < DMOD; ++j) mu += Hs[t][j];
        mu *= (1.0f / DMOD);
        float var = 0.f;
        for (int j = 0; j < DMOD; ++j) { float d = Hs[t][j] - mu; var += d * d; }
        var *= (1.0f / DMOD);
        const float inv = rsqrtf(var + 1e-5f);
        for (int j = 0; j < DMOD; ++j) {
            float h = (Hs[t][j] - mu) * inv * gs[j] + bs[j];
            Ah[t][j] = (_Float16)fmaxf(h, 0.f);
        }
    }
    __syncthreads();

    // ---- GEMM2: F[64,64] = Hn @ W2 + b2 ----
    {
        const int M = wave * 16 + mrow;
        const v8h* Arow = (const v8h*)&Ah[M][0];
        for (int n = 0; n < 4; ++n) {
            const int N = n * 16 + mrow;
            const v8h* Brow = (const v8h*)&W2t[N][0];
            v8f c = {};
            for (int kt = 0; kt < 4; ++kt) {
                v8h alo = Arow[kt * 4 + 2 * lh];
                v8h ahi = Arow[kt * 4 + 2 * lh + 1];
                v8h blo = Brow[kt * 4 + 2 * lh];
                v8h bhi = Brow[kt * 4 + 2 * lh + 1];
                v16h a, b;
#pragma unroll
                for (int e = 0; e < 8; ++e) {
                    a[e] = alo[e]; a[e + 8] = ahi[e];
                    b[e] = blo[e]; b[e + 8] = bhi[e];
                }
                c = __builtin_amdgcn_wmma_f32_16x16x32_f16(
                        false, a, false, b, (short)0, c, false, false);
            }
#pragma unroll
            for (int r = 0; r < 8; ++r) {
                const int Mo = wave * 16 + r + (lh << 3);
                const int No = n * 16 + mrow;
                feats[((size_t)g * SGRP + Mo) * DHALF + No] = c[r] + b2s[No];
            }
        }
    }
}

// ---------------------------------------------------------------------------
// Kernel 2: per-group FPS + comp-weighted soft k-means.  One 64-thread block
// (2 waves) per group; thread i = building i in group.  All data in LDS.
// ---------------------------------------------------------------------------
#define FPAD 68   // padded row (float4-aligned, bank-advance 4)

__global__ __launch_bounds__(64) void cluster_kernel(
    const float* __restrict__ comp,    // [8192,8192]
    const float* __restrict__ feats,   // [8192,64]
    const float* __restrict__ temp_p,
    int*  __restrict__ assign_out,     // [8192] global cluster ids (ws)
    float* __restrict__ out_assign)    // d_out[0:8192]
{
    __shared__ __align__(16) float F[SGRP][FPAD];
    __shared__ __align__(16) float Cm[SGRP][FPAD];
    __shared__ float Cent[KCL][DHALF];
    __shared__ float P[SGRP][KCL];
    __shared__ float red[SGRP];
    __shared__ int   redi[SGRP];
    __shared__ float colsum[KCL];
    __shared__ float cnt[KCL];
    __shared__ int   asg[SGRP];

    const int i = threadIdx.x;
    const int g = blockIdx.x;
    const float temp = temp_p[0];

    const float4* fr = (const float4*)(feats + ((size_t)g * SGRP + i) * DHALF);
    for (int d = 0; d < DHALF / 4; ++d) *(float4*)&F[i][d * 4] = fr[d];
    const float4* cr = (const float4*)(comp + ((size_t)g * SGRP + i) * NBLD + (size_t)g * SGRP);
    for (int j = 0; j < SGRP / 4; ++j) *(float4*)&Cm[i][j * 4] = cr[j];
    __syncthreads();

    // ---- FPS (first center = row 0, deterministic) ----
    Cent[0][i] = F[0][i];
    __syncthreads();
    float mind;
    {
        float d2 = 0.f;
        for (int d = 0; d < DHALF; ++d) { float df = F[i][d] - Cent[0][d]; d2 += df * df; }
        mind = sqrtf(fmaxf(d2, 0.f));
    }
    for (int c = 1; c < KCL; ++c) {
        red[i] = mind; redi[i] = i;
        __syncthreads();
        for (int s = 32; s > 0; s >>= 1) {
            if (i < s) {
                if (red[i + s] > red[i] ||
                    (red[i + s] == red[i] && redi[i + s] < redi[i])) {
                    red[i] = red[i + s]; redi[i] = redi[i + s];
                }
            }
            __syncthreads();
        }
        const int idx = redi[0];
        Cent[c][i] = F[idx][i];
        __syncthreads();
        float d2 = 0.f;
        for (int d = 0; d < DHALF; ++d) { float df = F[i][d] - Cent[c][d]; d2 += df * df; }
        mind = fminf(mind, sqrtf(fmaxf(d2, 0.f)));
        __syncthreads();
    }

    // ---- 10 Lloyd iterations with complementarity reweighting ----
    for (int it = 0; it < KM_ITERS; ++it) {
        float p[KCL];
        float mx = -3.4e38f;
        for (int k = 0; k < KCL; ++k) {
            float d2 = 0.f;
            for (int d = 0; d < DHALF; ++d) { float df = F[i][d] - Cent[k][d]; d2 += df * df; }
            p[k] = -sqrtf(fmaxf(d2, 0.f)) / temp;
            mx = fmaxf(mx, p[k]);
        }
        float s = 0.f;
        for (int k = 0; k < KCL; ++k) { p[k] = expf(p[k] - mx); s += p[k]; }
        for (int k = 0; k < KCL; ++k) { p[k] /= s; P[i][k] = p[k]; }
        __syncthreads();

        if (i < KCL) {
            float cs = 0.f;
            for (int j = 0; j < SGRP; ++j) cs += P[j][i];
            colsum[i] = cs;
        }
        __syncthreads();

        float s2 = 0.f;
        for (int k = 0; k < KCL; ++k) {
            float ac = 0.f;
            for (int j = 0; j < SGRP; ++j) ac += Cm[i][j] * P[j][k];
            ac /= (colsum[k] + EPSF);
            p[k] = P[i][k] * expf(-ac);
            s2 += p[k];
        }
        int a = 0; float best = -1.f;
        for (int k = 0; k < KCL; ++k) {
            p[k] = p[k] / (s2 + EPSF);
            if (p[k] > best) { best = p[k]; a = k; }
        }
        asg[i] = a;
        __syncthreads();

        if (i < KCL) {
            float cc = 0.f;
            for (int j = 0; j < SGRP; ++j) cc += (asg[j] == i) ? 1.f : 0.f;
            cnt[i] = cc;
        }
        __syncthreads();

        for (int k = 0; k < KCL; ++k) {
            if (cnt[k] > 0.f) {
                float acc = 0.f;
                for (int j = 0; j < SGRP; ++j)
                    if (asg[j] == k) acc += F[j][i];
                Cent[k][i] = acc / fmaxf(cnt[k], 1.0f);
            }
        }
        __syncthreads();
    }

    const int gid = asg[i] + g * KCL;
    assign_out[g * SGRP + i] = gid;
    out_assign[g * SGRP + i] = (float)gid;
}

// ---------------------------------------------------------------------------
// Kernel 3: all_probs [8192,8192] same-cluster indicator.  Block-diagonal:
// off-diagonal is a pure zero-fill.  Non-temporal B128 stores (write-once
// 256MB stream; keep it out of L2, assign array stays cached).
// ---------------------------------------------------------------------------
__global__ __launch_bounds__(256) void fill_probs(
    const int* __restrict__ assign, float* __restrict__ out)
{
    const size_t idx4 = ((size_t)blockIdx.x * 256 + threadIdx.x) * 4;
    const int i = (int)(idx4 >> 13);       // row
    const int j = (int)(idx4 & 8191);      // col (4-aligned, stays in one 64-block)
    v4f v;
    if ((i >> 6) != (j >> 6)) {
        v = (v4f){0.f, 0.f, 0.f, 0.f};
    } else {
        const int ai = assign[i];
        v[0] = (assign[j + 0] == ai) ? 1.f : 0.f;
        v[1] = (assign[j + 1] == ai) ? 1.f : 0.f;
        v[2] = (assign[j + 2] == ai) ? 1.f : 0.f;
        v[3] = (assign[j + 3] == ai) ? 1.f : 0.f;
    }
    __builtin_nontemporal_store(v, (v4f*)(out + idx4));
}

// ---------------------------------------------------------------------------
extern "C" void kernel_launch(void* const* d_in, const int* in_sizes, int n_in,
                              void* d_out, int out_size, void* d_ws, size_t ws_size,
                              hipStream_t stream) {
    const float* X    = (const float*)d_in[0];  // embeddings [1,8192,128]
    const float* comp = (const float*)d_in[1];  // [8192,8192]
    //                 d_in[2] lv_group_ids (implicit: arange(N)//64)
    const float* W1   = (const float*)d_in[3];
    const float* b1   = (const float*)d_in[4];
    const float* gma  = (const float*)d_in[5];
    const float* bta  = (const float*)d_in[6];
    const float* W2   = (const float*)d_in[7];
    const float* b2   = (const float*)d_in[8];
    const float* temp = (const float*)d_in[9];

    float* feats  = (float*)d_ws;                                   // 2 MB
    int*   assign = (int*)((char*)d_ws + (size_t)NBLD * DHALF * 4); // 32 KB
    float* out    = (float*)d_out;

    enc_kernel<<<NGRP, 128, 0, stream>>>(X, W1, b1, gma, bta, W2, b2, feats);
    cluster_kernel<<<NGRP, SGRP, 0, stream>>>(comp, feats, temp, assign, out);

    const size_t nq = (size_t)NBLD * NBLD / 4;      // float4 stores
    fill_probs<<<(unsigned)(nq / 256), 256, 0, stream>>>(assign, out + NBLD);
}
